// DeformationNetwork_7791070675874
// MI455X (gfx1250) — matrix operand order repaired and checked by reading, not compiled, not run on hardware
//
#include <hip/hip_runtime.h>
#include <hip/hip_bf16.h>

typedef __attribute__((ext_vector_type(16))) _Float16 v16h;
typedef __attribute__((ext_vector_type(8)))  float    v8f;
typedef __attribute__((ext_vector_type(4)))  unsigned int u32x4;
typedef __attribute__((ext_vector_type(8)))  int      i32x8;
typedef __attribute__((ext_vector_type(4)))  int      i32x4;

#define HIDDEN    256
#define ROWS_PER_BLOCK 128
#define PI_F      3.14159265358979f

// f16 weight workspace layout (element offsets into wsW):
//  W0p  [256][64]   @ 0        (16384)
//  W1p  [256][256]  @ 16384    (65536)
//  W2p  [256][320]  @ 81920    (81920)   (W2a = rows 0..127, W2b = rows 128..255)
//  W3p  [256][256]  @ 163840   (65536)   (W3a / W3b halves)
//  Whp  [16][256]   @ 229376   (4096)
#define OFS_W0 0
#define OFS_W1 16384
#define OFS_W2 81920
#define OFS_W3 163840
#define OFS_WH 229376
#define TOTAL_WH 233472
#define TOTAL_B 1040

// LDS byte offsets (dynamic smem), total 299072 <= 320KB:
//  regionB @ 0       (131072 B)  : W1, W2b, W3b
//  regionA @ 131072  (81920 B)   : W0, W2a, W3a, Wh
//  sX      @ 212992  (16384 B)   : 128 x 64 f16
//  sH      @ 229376  (65536 B)   : 128 x 256 f16
//  sBias   @ 294912  (4160 B)
#define LDS_B   0
#define LDS_A   131072
#define LDS_X   212992
#define LDS_H   229376
#define LDS_BIAS 294912
#define LDS_TOTAL 299072

#if defined(__has_builtin)
#if __has_builtin(__builtin_amdgcn_tensor_load_to_lds)
#define HAVE_TDM 1
#endif
#endif
#ifndef HAVE_TDM
#define HAVE_TDM 0
#endif

union FragU { v16h v; u32x4 q[2]; };

__device__ __forceinline__ v16h load_frag16(const _Float16* p) {
  FragU f;
  f.q[0] = *(const u32x4*)p;
  f.q[1] = *(const u32x4*)(p + 8);
  return f.v;
}

// A fragment (16x32 f16): chunk0 (VGPR0-3) K = kt*32 + half*8 ; chunk1 (VGPR4-7) K = kt*32 + 16 + half*8
__device__ __forceinline__ v16h load_a(const _Float16* base, int row, int kpad, int kt, int hsel) {
  const _Float16* p = base + row * kpad + kt * 32 + hsel * 8;
  FragU f;
  f.q[0] = *(const u32x4*)p;
  f.q[1] = *(const u32x4*)(p + 16);
  return f.v;
}

// B fragment (32x16 f16): lane holds out col n = lane&15; chunk0 K = kt*32 + half*8 ; chunk1 K = kt*32 + 16 + half*8
__device__ __forceinline__ v16h load_b(const _Float16* w, int wrow, int kpad, int kt, int hsel) {
  const _Float16* p = w + wrow * kpad + kt * 32 + hsel * 8;
  FragU f;
  f.q[0] = *(const u32x4*)p;
  f.q[1] = *(const u32x4*)(p + 16);
  return f.v;
}

__device__ __forceinline__ void copy_w(_Float16* dst, const _Float16* src, unsigned bytes, int tid) {
  const u32x4* s = (const u32x4*)src;
  u32x4* d = (u32x4*)dst;
  const int n = (int)(bytes >> 4);
  for (int i = tid; i < n; i += 256) d[i] = s[i];
}

#if HAVE_TDM
// Tensor Data Mover: contiguous copy expressed as a 2D tile of (rows x 64) 8-byte units.
// D# per CDNA5 ISA. bytes must be a multiple of 512.
__device__ __forceinline__ void tdm_load(void* ldsDst, const void* gsrc, unsigned bytes) {
  const unsigned rows = bytes >> 9;                        // 512B per row
  const unsigned long long ga = (unsigned long long)(uintptr_t)gsrc;
  const unsigned lds = (unsigned)(uintptr_t)ldsDst;        // low 32 bits = LDS byte offset
  u32x4 g0;
  g0.x = 1u;                                               // count=1, no gather, user mode
  g0.y = lds;                                              // lds_addr
  g0.z = (unsigned)ga;                                     // global_addr[31:0]
  g0.w = (unsigned)((ga >> 32) & 0x01FFFFFFu) | (2u << 30);// global_addr[56:32] | type=2
  i32x8 g1;
  g1[0] = (int)(3u << 16);          // workgroup_mask=0, data_size=3 (8B), no flags
  g1[1] = (int)(64u << 16);         // tensor_dim0 low16 = 64
  g1[2] = (int)(rows << 16);        // tensor_dim0 hi = 0 | tensor_dim1 low16 = rows
  g1[3] = (int)(64u << 16);         // tensor_dim1 hi = 0 | tile_dim0 = 64
  g1[4] = (int)rows;                // tile_dim1 = rows, tile_dim2 = 0
  g1[5] = 64;                       // tensor_dim0_stride low32 = 64
  g1[6] = (int)(64u << 16);         // stride0 hi = 0 | tensor_dim1_stride low16 = 64
  g1[7] = 0;
  i32x4 z4 = {0, 0, 0, 0};
#if __has_include(<hip/amd_detail/amd_gfx1250_TDM.h>)
  i32x8 z8 = {0, 0, 0, 0, 0, 0, 0, 0};
  __builtin_amdgcn_tensor_load_to_lds(g0, g1, z4, z4, z8, 0);
#else
  __builtin_amdgcn_tensor_load_to_lds(g0, g1, z4, z4, 0);
#endif
}
#endif

// Issue staging of one weight chunk (TDM from wave 0, or cooperative fallback).
__device__ __forceinline__ void stage_weights(_Float16* dst, const _Float16* src,
                                              unsigned bytes, int tid) {
#if HAVE_TDM
  if (tid < 32) tdm_load(dst, src, bytes);
#else
  copy_w(dst, src, bytes, tid);
#endif
}

__device__ __forceinline__ void weights_fence(int tid) {
#if HAVE_TDM
  if (tid < 32) __builtin_amdgcn_s_wait_tensorcnt(0);
#endif
}

// Two independent output tiles (nt, nt+1) per iteration: interleaved WMMA chains hide
// XDL latency, and 4 ds_load_b128 issue ahead of the first consuming WMMA so the
// dscnt wait is partial instead of 0.
template <int KT>
__device__ __forceinline__ void dense_pair(const _Float16* sW, int kpad, const v16h* a,
                                           const float* bias, _Float16* sHout,
                                           int lane, int waveRow, bool relu,
                                           int ntBegin, int ntEnd, int wrowBase) {
  const int n16  = lane & 15;
  const int hsel = lane >> 4;
  for (int nt = ntBegin; nt < ntEnd; nt += 2) {
    const int wr0 = nt * 16 + n16 - wrowBase;
    const int wr1 = wr0 + 16;
    const float bv0 = bias[nt * 16 + n16];
    const float bv1 = bias[nt * 16 + 16 + n16];
    v8f c0 = {}, c1 = {};
#pragma unroll
    for (int kt = 0; kt < KT; ++kt) {
      v16h b0 = load_b(sW, wr0, kpad, kt, hsel);
      v16h b1 = load_b(sW, wr1, kpad, kt, hsel);
      c0 = __builtin_amdgcn_wmma_f32_16x16x32_f16(false, a[kt], false, b0,
                                                  (short)0, c0, false, false);
      c1 = __builtin_amdgcn_wmma_f32_16x16x32_f16(false, a[kt], false, b1,
                                                  (short)0, c1, false, false);
    }
#pragma unroll
    for (int r = 0; r < 8; ++r) {
      float x0 = c0[r] + bv0;
      float x1 = c1[r] + bv1;
      if (relu) { x0 = fmaxf(x0, 0.0f); x1 = fmaxf(x1, 0.0f); }
      const int ofs = (waveRow + hsel * 8 + r) * HIDDEN + nt * 16 + n16;
      sHout[ofs]      = (_Float16)x0;
      sHout[ofs + 16] = (_Float16)x1;
    }
  }
}

// ---------------- prep kernel: f32 weights -> padded f16 workspace ----------------
__global__ void deform_prep(const float* __restrict__ W0, const float* __restrict__ W1,
                            const float* __restrict__ W2, const float* __restrict__ W3,
                            const float* __restrict__ Wxyz, const float* __restrict__ Wrot,
                            const float* __restrict__ Wscale, const float* __restrict__ Wop,
                            const float* __restrict__ b0, const float* __restrict__ b1,
                            const float* __restrict__ b2, const float* __restrict__ b3,
                            const float* __restrict__ bxyz, const float* __restrict__ brot,
                            const float* __restrict__ bscale, const float* __restrict__ bop,
                            _Float16* __restrict__ wsW, float* __restrict__ wsB) {
  int gid = blockIdx.x * 256 + threadIdx.x;
  if (gid < TOTAL_WH) {
    float v = 0.0f;
    if (gid < OFS_W1) {
      int r = gid >> 6, c = gid & 63;
      if (c < 60) v = W0[r * 60 + c];
    } else if (gid < OFS_W2) {
      v = W1[gid - OFS_W1];
    } else if (gid < OFS_W3) {
      int i = gid - OFS_W2;
      int r = i / 320, c = i % 320;
      if (c < 316) v = W2[r * 316 + c];
    } else if (gid < OFS_WH) {
      v = W3[gid - OFS_W3];
    } else {
      int i = gid - OFS_WH;
      int r = i >> 8, c = i & 255;
      if (r < 3)        v = Wxyz[r * 256 + c];
      else if (r < 7)   v = Wrot[(r - 3) * 256 + c];
      else if (r < 10)  v = Wscale[(r - 7) * 256 + c];
      else if (r == 10) v = Wop[c];
    }
    wsW[gid] = (_Float16)v;
  } else if (gid < TOTAL_WH + TOTAL_B) {
    int j = gid - TOTAL_WH;
    float v = 0.0f;
    if (j < 256)       v = b0[j];
    else if (j < 512)  v = b1[j - 256];
    else if (j < 768)  v = b2[j - 512];
    else if (j < 1024) v = b3[j - 768];
    else {
      int k = j - 1024;
      if (k < 3)        v = bxyz[k];
      else if (k < 7)   v = brot[k - 3];
      else if (k < 10)  v = bscale[k - 7];
      else if (k == 10) v = bop[0];
    }
    wsB[j] = v;
  }
}

// ---------------- main kernel: TDM-pipelined encode + 4-layer MLP + heads ----------------
__global__ void __launch_bounds__(256)
deform_main(const float* __restrict__ xyz, const float* __restrict__ timep,
            const _Float16* __restrict__ wsW, const float* __restrict__ wsB,
            float* __restrict__ out, int nPoints) {
  extern __shared__ char smem[];
  _Float16* rB    = (_Float16*)(smem + LDS_B);
  _Float16* rA    = (_Float16*)(smem + LDS_A);
  _Float16* sX    = (_Float16*)(smem + LDS_X);
  _Float16* sH    = (_Float16*)(smem + LDS_H);
  float*    sBias = (float*)(smem + LDS_BIAS);

  const int tid  = threadIdx.x;
  const int lane = tid & 31;
  const int wave = tid >> 5;
  const int blockBase = blockIdx.x * ROWS_PER_BLOCK;

  // Phase -1: kick W0 DMA immediately, overlap with encode + bias staging.
  stage_weights(rA, wsW + OFS_W0, 32768, tid);

  for (int j = tid; j < TOTAL_B; j += 256) sBias[j] = wsB[j];

  if (tid < ROWS_PER_BLOCK) {
    const int prow = blockBase + tid;
    _Float16* dst = sX + tid * 64;
    const float px = xyz[prow * 3 + 0];
    const float py = xyz[prow * 3 + 1];
    const float pz = xyz[prow * 3 + 2];
    const float t  = timep[0];
    dst[0] = (_Float16)px; dst[1] = (_Float16)py; dst[2] = (_Float16)pz;
    float p3[3] = {px, py, pz};
#pragma unroll
    for (int d = 0; d < 3; ++d) {
      float s = p3[d] * PI_F;
#pragma unroll
      for (int f = 0; f < 6; ++f) {
        float ang = s * (float)(1 << f);
        dst[3 + 12 * d + f]     = (_Float16)__sinf(ang);
        dst[3 + 12 * d + 6 + f] = (_Float16)__cosf(ang);
      }
    }
    dst[39] = (_Float16)t;
    float ts = t * PI_F;
#pragma unroll
    for (int f = 0; f < 10; ++f) {
      float ang = ts * (float)(1 << f);
      dst[40 + f] = (_Float16)__sinf(ang);
      dst[50 + f] = (_Float16)__cosf(ang);
    }
    dst[60] = (_Float16)0.f; dst[61] = (_Float16)0.f;
    dst[62] = (_Float16)0.f; dst[63] = (_Float16)0.f;
  }

  weights_fence(tid);
  __syncthreads();                                   // W0 in A, sX ready

  const int waveRow = wave * 16;
  const int mrow    = waveRow + (lane & 15);
  const int hsel    = lane >> 4;
  v16h a[10];

  // L0 (A) || TDM W1 -> B
  stage_weights(rB, wsW + OFS_W1, 131072, tid);
  a[0] = load_a(sX, mrow, 64, 0, hsel);
  a[1] = load_a(sX, mrow, 64, 1, hsel);
  dense_pair<2>(rA, 64, a, sBias + 0, sH, lane, waveRow, true, 0, 16, 0);
  weights_fence(tid);
  __syncthreads();                                   // A free, W1 ready

  // L1 (B) || TDM W2a -> A
  stage_weights(rA, wsW + OFS_W2, 81920, tid);
#pragma unroll
  for (int kt = 0; kt < 8; ++kt) a[kt] = load_a(sH, mrow, 256, kt, hsel);
  dense_pair<8>(rB, 256, a, sBias + 256, sH, lane, waveRow, true, 0, 16, 0);
  weights_fence(tid);
  __syncthreads();                                   // B free, W2a ready

  // L2 cols 0..127 (A) || TDM W2b -> B
  stage_weights(rB, wsW + OFS_W2 + 40960, 81920, tid);
#pragma unroll
  for (int kt = 0; kt < 8; ++kt) a[kt] = load_a(sH, mrow, 256, kt, hsel);
  a[8] = load_a(sX, mrow, 64, 0, hsel);
  a[9] = load_a(sX, mrow, 64, 1, hsel);
  dense_pair<10>(rA, 320, a, sBias + 512, sH, lane, waveRow, true, 0, 8, 0);
  weights_fence(tid);
  __syncthreads();                                   // A free, W2b ready

  // L2 cols 128..255 (B) || TDM W3a -> A   (a[] reused across the barrier)
  stage_weights(rA, wsW + OFS_W3, 65536, tid);
  dense_pair<10>(rB, 320, a, sBias + 512, sH, lane, waveRow, true, 8, 16, 128);
  weights_fence(tid);
  __syncthreads();                                   // B free, W3a ready

  // L3 cols 0..127 (A) || TDM W3b -> B
  stage_weights(rB, wsW + OFS_W3 + 32768, 65536, tid);
#pragma unroll
  for (int kt = 0; kt < 8; ++kt) a[kt] = load_a(sH, mrow, 256, kt, hsel);
  dense_pair<8>(rA, 256, a, sBias + 768, sH, lane, waveRow, true, 0, 8, 0);
  weights_fence(tid);
  __syncthreads();                                   // A free, W3b ready

  // L3 cols 128..255 (B) || TDM Wh -> A
  stage_weights(rA, wsW + OFS_WH, 8192, tid);
  dense_pair<8>(rB, 256, a, sBias + 768, sH, lane, waveRow, true, 8, 16, 128);
  weights_fence(tid);
  __syncthreads();                                   // B free, Wh ready

  // heads (A): 256 -> 16 (11 valid)
#pragma unroll
  for (int kt = 0; kt < 8; ++kt) a[kt] = load_a(sH, mrow, 256, kt, hsel);
  v8f c = {};
#pragma unroll
  for (int kt = 0; kt < 8; ++kt) {
    v16h b = load_b(rA, (lane & 15), 256, kt, hsel);
    c = __builtin_amdgcn_wmma_f32_16x16x32_f16(false, a[kt], false, b,
                                               (short)0, c, false, false);
  }
  const int j = lane & 15;
  const float bv = sBias[1024 + j];
#pragma unroll
  for (int r = 0; r < 8; ++r) {
    const int row = blockBase + waveRow + hsel * 8 + r;
    const float v = c[r] + bv;
    if (j < 3)        out[row * 3 + j] = v;                      // delta_xyz
    else if (j < 7)   out[nPoints * 3 + row * 4 + (j - 3)] = v;  // delta_rotation
    else if (j < 10)  out[nPoints * 7 + row * 3 + (j - 7)] = v;  // delta_scaling
    else if (j == 10) out[nPoints * 10 + row] = v;               // delta_opacity
  }
}

extern "C" void kernel_launch(void* const* d_in, const int* in_sizes, int n_in,
                              void* d_out, int out_size, void* d_ws, size_t ws_size,
                              hipStream_t stream) {
  const float* xyz    = (const float*)d_in[0];
  const float* timep  = (const float*)d_in[1];
  const float* W0     = (const float*)d_in[2];
  const float* b0     = (const float*)d_in[3];
  const float* W1     = (const float*)d_in[4];
  const float* b1     = (const float*)d_in[5];
  const float* W2     = (const float*)d_in[6];
  const float* b2     = (const float*)d_in[7];
  const float* W3     = (const float*)d_in[8];
  const float* b3     = (const float*)d_in[9];
  const float* Wxyz   = (const float*)d_in[10];
  const float* bxyz   = (const float*)d_in[11];
  const float* Wrot   = (const float*)d_in[12];
  const float* brot   = (const float*)d_in[13];
  const float* Wscale = (const float*)d_in[14];
  const float* bscale = (const float*)d_in[15];
  const float* Wop    = (const float*)d_in[16];
  const float* bop    = (const float*)d_in[17];

  float*    wsB = (float*)d_ws;                        // 1040 f32
  _Float16* wsW = (_Float16*)((char*)d_ws + 8192);     // 233472 f16

  const int prepThreads = TOTAL_WH + TOTAL_B;
  deform_prep<<<(prepThreads + 255) / 256, 256, 0, stream>>>(
      W0, W1, W2, W3, Wxyz, Wrot, Wscale, Wop,
      b0, b1, b2, b3, bxyz, brot, bscale, bop, wsW, wsB);

  const int nPoints = in_sizes[0] / 3;                 // 262144
  const int grid = nPoints / ROWS_PER_BLOCK;           // 2048 blocks of 8 wave32s
  deform_main<<<grid, 256, LDS_TOTAL, stream>>>(xyz, timep, wsW, wsB,
                                                (float*)d_out, nPoints);
}
